// RNNSeqEncoder_42709154791830
// MI455X (gfx1250) — matrix operand order repaired and checked by reading, not compiled, not run on hardware
//
#include <hip/hip_runtime.h>
#include <hip/hip_bf16.h>

// GRU dims (fixed by the reference)
constexpr int B = 64, S = 512, D = 512, H = 512;
// 128 WGs: one 16x16 output tile each (4 row-blocks x 32 col-blocks).
// 8 waves per WG: K=512 split into 8 slices of 64 (weights register-resident).
constexpr int NUM_WG = 128;
constexpr int WG_THREADS = 256;

typedef __attribute__((ext_vector_type(16))) __bf16 v16bf;
typedef __attribute__((ext_vector_type(8)))  float  v8f;

union Frag16 {           // 16 bf16 = 8 VGPRs, loaded as two 16-byte chunks
    v16bf v;
    uint4 u[2];
};

__device__ __forceinline__ unsigned short f32_to_bf16_rne(float f) {
    unsigned int u = __float_as_uint(f);
    u += 0x7FFFu + ((u >> 16) & 1u);   // round-to-nearest-even
    return (unsigned short)(u >> 16);
}

// ---------------------------------------------------------------------------
// f32 -> bf16 bulk convert
// ---------------------------------------------------------------------------
__global__ void cvt_f32_bf16_kernel(const float* __restrict__ in,
                                    unsigned short* __restrict__ out, int n) {
    int i = blockIdx.x * blockDim.x + threadIdx.x;
    if (i < n) out[i] = f32_to_bf16_rne(in[i]);
}

// ---------------------------------------------------------------------------
// zero h ping/pong buffers (f32 + bf16) and the grid-barrier counter
// ---------------------------------------------------------------------------
__global__ void gru_init_kernel(float* __restrict__ hf, unsigned short* __restrict__ hb,
                                unsigned int* __restrict__ bar) {
    int i = blockIdx.x * blockDim.x + threadIdx.x;
    if (i < 2 * B * H) { hf[i] = 0.0f; hb[i] = 0; }
    if (i == 0) *bar = 0u;
}

// ---------------------------------------------------------------------------
// Persistent fused GRU scan (persistent-RNN style).
//   - WG (256 thr, 8 waves) owns one 16x16 (batch-rows x hidden-cols) tile.
//   - Wave w handles K-slice [w*64, w*64+64): its 12 weight fragments
//     (2 ks x 3 gates x {W_ih,W_hh}) are loaded ONCE and live in VGPRs for
//     all 512 timesteps -> per-step inner loop is A-loads + 12 WMMAs only.
//   - Partial sums reduced through LDS (4 arrays: r-sum, z-sum, xn, ghn);
//     epilogue: 256 threads x 1 output element.
//   - Device-wide sense barrier between steps (128 WGs co-resident).
// ---------------------------------------------------------------------------
__global__ __launch_bounds__(WG_THREADS, 1)
void gru_scan_kernel(const float* __restrict__ x_f32,          // [B,S,D]
                     const unsigned short* __restrict__ x_bf,  // [B,S,D] bf16
                     const unsigned short* __restrict__ Wih,   // [3H,D] bf16
                     const unsigned short* __restrict__ Whh,   // [3H,H] bf16
                     const float* __restrict__ b_ih,           // [3H]
                     const float* __restrict__ b_hh,           // [3H]
                     float* __restrict__ hf,                   // [2][B,H] f32 ping/pong
                     unsigned short* __restrict__ hb,          // [2][B,H] bf16 ping/pong
                     float* __restrict__ out_res,              // [B,S,H]
                     float* __restrict__ out_hid,              // [B,S,H]
                     unsigned int* __restrict__ bar) {
    // lds[wave][gatePart][lane][v]  gateParts: 0=r-sum 1=z-sum 2=xn 3=ghn
    __shared__ float lds[8][4][32][8];    // 32 KB

    const int tid   = threadIdx.x;
    const int lane  = tid & 31;
    const int wave  = tid >> 5;           // 0..7 -> K-slice
    const int r0    = (blockIdx.x >> 5) * 16;   // batch-row tile base
    const int c0    = (blockIdx.x & 31) * 16;   // hidden-col tile base

    const int l16   = lane & 15;
    const int lhalf = lane >> 4;
    const int kwb   = wave * 64;          // this wave's K base

    // ---- one-time: preload this wave's 12 weight fragments into VGPRs ----
    // B = W^T: column n of B is row (g*H + c0 + n) of W -> contiguous in memory.
    // B layout: lane holds col = lane%16, K = (lane/16)*16 .. +15 contiguous.
    Frag16 wfi[2][3], wfh[2][3];
#pragma unroll
    for (int ks = 0; ks < 2; ++ks) {
#pragma unroll
        for (int g = 0; g < 3; ++g) {
            const size_t wrow = (size_t)(g * H + c0 + l16);
            const unsigned short* p = Wih + wrow * D + kwb + ks * 32 + lhalf * 16;
            wfi[ks][g].u[0] = *(const uint4*)(p);
            wfi[ks][g].u[1] = *(const uint4*)(p + 8);
            const unsigned short* q = Whh + wrow * H + kwb + ks * 32 + lhalf * 16;
            wfh[ks][g].u[0] = *(const uint4*)(q);
            wfh[ks][g].u[1] = *(const uint4*)(q + 8);
        }
    }

    // ---- epilogue-thread mapping: tid -> one element of the 16x16 tile ----
    const int em   = tid >> 4;            // 0..15 tile row
    const int en   = tid & 15;            // 0..15 tile col
    const int erow = r0 + em;             // batch index
    const int ecol = c0 + en;             // hidden index
    const int rl   = (em >> 3) * 16 + en; // lane holding this elem in C layout
    const int rv   = em & 7;              // VGPR index holding this elem

    const float bir = b_ih[ecol],         bhr = b_hh[ecol];
    const float biz = b_ih[H + ecol],     bhz = b_hh[H + ecol];
    const float bin = b_ih[2 * H + ecol], bhn = b_hh[2 * H + ecol];

    unsigned int barTarget = 0;

    for (int t = 0; t < S; ++t) {
        const unsigned short* h_rd_b = hb + (size_t)(t & 1) * (B * H);
        const float*          h_rd_f = hf + (size_t)(t & 1) * (B * H);
        unsigned short*       h_wr_b = hb + (size_t)((t + 1) & 1) * (B * H);
        float*                h_wr_f = hf + (size_t)((t + 1) & 1) * (B * H);

        // ---- GEMM phase: 2 A-load pairs + 12 WMMAs, weights from VGPRs ----
        v8f accx[3] = {v8f{}, v8f{}, v8f{}};
        v8f acch[3] = {v8f{}, v8f{}, v8f{}};
#pragma unroll
        for (int ks = 0; ks < 2; ++ks) {
            const int kb = kwb + ks * 32;
            // A layout: lane holds M = lane%16, K = (lane/16)*16 .. +15 contiguous
            Frag16 ax, ah;
            const unsigned short* xp =
                x_bf + ((size_t)(r0 + l16) * S + t) * D + kb + lhalf * 16;
            ax.u[0] = *(const uint4*)(xp);
            ax.u[1] = *(const uint4*)(xp + 8);
            const unsigned short* hp = h_rd_b + (size_t)(r0 + l16) * H + kb + lhalf * 16;
            ah.u[0] = *(const uint4*)(hp);
            ah.u[1] = *(const uint4*)(hp + 8);
#pragma unroll
            for (int g = 0; g < 3; ++g) {
                accx[g] = __builtin_amdgcn_wmma_f32_16x16x32_bf16(
                    false, ax.v, false, wfi[ks][g].v, (short)0, accx[g], false, false);
                acch[g] = __builtin_amdgcn_wmma_f32_16x16x32_bf16(
                    false, ah.v, false, wfh[ks][g].v, (short)0, acch[g], false, false);
            }
        }

        // ---- partial sums -> LDS (r and z only need xp+gh; n needs both) ----
        {
            v8f sr = accx[0] + acch[0];
            v8f sz = accx[1] + acch[1];
            *(v8f*)&lds[wave][0][lane][0] = sr;
            *(v8f*)&lds[wave][1][lane][0] = sz;
            *(v8f*)&lds[wave][2][lane][0] = accx[2];
            *(v8f*)&lds[wave][3][lane][0] = acch[2];
        }
        __syncthreads();

        // ---- epilogue: each thread reduces 8 waves and emits one element ----
        {
            float s_r = 0.f, s_z = 0.f, s_xn = 0.f, s_gn = 0.f;
#pragma unroll
            for (int w = 0; w < 8; ++w) {
                s_r  += lds[w][0][rl][rv];
                s_z  += lds[w][1][rl][rv];
                s_xn += lds[w][2][rl][rv];
                s_gn += lds[w][3][rl][rv];
            }
            const float gr = 1.0f / (1.0f + __expf(-(s_r + bir + bhr)));
            const float gz = 1.0f / (1.0f + __expf(-(s_z + biz + bhz)));
            const float gn = tanhf(s_xn + bin + gr * (s_gn + bhn));
            const float hold = h_rd_f[(size_t)erow * H + ecol];
            const float hnew = (1.0f - gz) * gn + gz * hold;

            h_wr_f[(size_t)erow * H + ecol] = hnew;
            h_wr_b[(size_t)erow * H + ecol] = f32_to_bf16_rne(hnew);

            const size_t oi = ((size_t)erow * S + t) * H + ecol;  // D==H: same flat idx
            out_res[oi] = x_f32[oi] + hnew;
            out_hid[oi] = hnew;
        }

        // ---- device-wide barrier (sense via monotonically increasing count) ----
        __threadfence();                 // release h_wr stores at agent scope
        __syncthreads();                 // LDS reuse safety + all writes issued
        if (tid == 0) {
            __hip_atomic_fetch_add(bar, 1u, __ATOMIC_RELEASE, __HIP_MEMORY_SCOPE_AGENT);
        }
        barTarget += NUM_WG;
        if (tid == 0) {
            while (__hip_atomic_load(bar, __ATOMIC_ACQUIRE, __HIP_MEMORY_SCOPE_AGENT) < barTarget) {
                __builtin_amdgcn_s_sleep(1);
            }
        }
        __syncthreads();
        __threadfence();                 // acquire: invalidate stale h lines
    }
}

// ---------------------------------------------------------------------------
// Host side
// ---------------------------------------------------------------------------
extern "C" void kernel_launch(void* const* d_in, const int* in_sizes, int n_in,
                              void* d_out, int out_size, void* d_ws, size_t ws_size,
                              hipStream_t stream) {
    (void)in_sizes; (void)n_in; (void)out_size; (void)ws_size;

    const float* x_f32 = (const float*)d_in[0];   // [B,S,D]
    const float* Wih_f = (const float*)d_in[1];   // [3H,D]
    const float* Whh_f = (const float*)d_in[2];   // [3H,H]
    const float* b_ih  = (const float*)d_in[3];   // [3H]
    const float* b_hh  = (const float*)d_in[4];   // [3H]

    // Workspace layout (all region sizes are multiples of 256 B)
    char* ws = (char*)d_ws;
    size_t off = 0;
    unsigned short* x_bf  = (unsigned short*)(ws + off); off += (size_t)B * S * D * 2; // 32 MB
    unsigned short* Wih_b = (unsigned short*)(ws + off); off += (size_t)3 * H * D * 2; // 1.5 MB
    unsigned short* Whh_b = (unsigned short*)(ws + off); off += (size_t)3 * H * H * 2; // 1.5 MB
    float*          hf    = (float*)         (ws + off); off += (size_t)2 * B * H * 4; // ping/pong
    unsigned short* hb    = (unsigned short*)(ws + off); off += (size_t)2 * B * H * 2;
    unsigned int*   bar   = (unsigned int*)  (ws + off);

    float* out_res = (float*)d_out;
    float* out_hid = out_res + (size_t)B * S * H;

    // 1) bf16 conversions (one-shot, fully parallel)
    {
        int n = B * S * D;
        cvt_f32_bf16_kernel<<<(n + 255) / 256, 256, 0, stream>>>(x_f32, x_bf, n);
        n = 3 * H * D;
        cvt_f32_bf16_kernel<<<(n + 255) / 256, 256, 0, stream>>>(Wih_f, Wih_b, n);
        n = 3 * H * H;
        cvt_f32_bf16_kernel<<<(n + 255) / 256, 256, 0, stream>>>(Whh_f, Whh_b, n);
    }

    // 2) zero h0 (both buffers) + barrier counter
    {
        int n = 2 * B * H;
        gru_init_kernel<<<(n + 255) / 256, 256, 0, stream>>>(hf, hb, bar);
    }

    // 3) persistent fused GRU scan (single launch, internal t-loop + grid barrier)
    gru_scan_kernel<<<NUM_WG, WG_THREADS, 0, stream>>>(
        x_f32, x_bf, Wih_b, Whh_b, b_ih, b_hh, hf, hb, out_res, out_hid, bar);
}
